// GPT2FrozenStateFusion_62534723830144
// MI455X (gfx1250) — matrix operand order repaired and checked by reading, not compiled, not run on hardware
//
#include <hip/hip_runtime.h>

// ---------------------------------------------------------------------------
// GPT-2 small forward (B=8,T=1024,D=768,12 layers) + state fusion @ layer 8.
// All large matmuls run through V_WMMA_F32_16X16X32_BF16 (f32 accumulate).
// Weights converted fp32 -> bf16 transposed [N][K] on device each call.
// ---------------------------------------------------------------------------

typedef __attribute__((ext_vector_type(16))) __bf16 v16bf;
typedef __attribute__((ext_vector_type(8)))  float  v8f;
typedef unsigned short bh;   // raw bf16 storage (trivial type, safe in LDS)

__device__ __forceinline__ bh f2bf(float f) {
    unsigned int u = __float_as_uint(f);
    unsigned int r = (u + 0x7FFFu + ((u >> 16) & 1u)) >> 16;
    return (bh)r;
}

union Frag16 { v16bf v; uint4 u[2]; };

__device__ __forceinline__ void ld_frag(Frag16& f, const bh* lo, const bh* hi) {
    f.u[0] = *reinterpret_cast<const uint4*>(lo);
    f.u[1] = *reinterpret_cast<const uint4*>(hi);
}

__device__ __forceinline__ v8f wmma_bf16(const v16bf& a, const v16bf& b, const v8f& c) {
    return __builtin_amdgcn_wmma_f32_16x16x32_bf16(false, a, false, b, (short)0, c,
                                                   false, false);
}

__device__ __forceinline__ v8f vzero8() {
    v8f z;
#pragma unroll
    for (int i = 0; i < 8; ++i) z[i] = 0.0f;
    return z;
}

// ---------------------------------------------------------------------------
// Sequential group scan: s_t = mul[x_t, s_{t-1}]
// ---------------------------------------------------------------------------
__global__ void scan_kernel(const int* __restrict__ ids, const int* __restrict__ mul,
                            int* __restrict__ sid) {
    int b = threadIdx.x;
    if (b < 8) {
        int s = 0;
        for (int t = 0; t < 1024; ++t) {
            int x = ids[b * 1024 + t];
            s = mul[x * 60 + s];
            sid[b * 1024 + t] = s;
        }
    }
}

// ---------------------------------------------------------------------------
// Weight convert + transpose: in fp32 [K][N] -> out bf16 [N][K]
// ---------------------------------------------------------------------------
__global__ __launch_bounds__(256) void conv_w_kernel(const float* __restrict__ in,
                                                     bh* __restrict__ out,
                                                     int K, int N) {
    __shared__ float tile[32][33];
    int n0 = blockIdx.x * 32, k0 = blockIdx.y * 32;
    int tid = threadIdx.x;
    int cn = tid & 31, rk = tid >> 5;
#pragma unroll
    for (int p = 0; p < 4; ++p)
        tile[rk + p * 8][cn] = in[(size_t)(k0 + rk + p * 8) * N + n0 + cn];
    __syncthreads();
    int ck = tid & 31, rn = tid >> 5;
#pragma unroll
    for (int p = 0; p < 4; ++p)
        out[(size_t)(n0 + rn + p * 8) * K + k0 + ck] = f2bf(tile[ck][rn + p * 8]);
}

// ---------------------------------------------------------------------------
// Embedding: h[b,t,:] = tok_emb[id] + wpe[t]
// ---------------------------------------------------------------------------
__global__ __launch_bounds__(256) void embed_kernel(const int* __restrict__ ids,
                                                    const float* __restrict__ tok,
                                                    const float* __restrict__ wpe,
                                                    float* __restrict__ h) {
    int bt = blockIdx.x;
    int t = bt & 1023;
    int id = ids[bt];
    for (int d = threadIdx.x; d < 768; d += 256)
        h[(size_t)bt * 768 + d] = tok[id * 768 + d] + wpe[t * 768 + d];
}

// Gather state_emb rows -> bf16 [8192,128]
__global__ __launch_bounds__(256) void seg_kernel(const int* __restrict__ sid,
                                                  const float* __restrict__ se,
                                                  bh* __restrict__ seg) {
    int i = blockIdx.x * 256 + threadIdx.x;
    int row = i >> 7, c = i & 127;
    seg[i] = f2bf(se[sid[row] * 128 + c]);
}

// fp32 -> bf16 elementwise
__global__ __launch_bounds__(256) void cvt_kernel(const float* __restrict__ x,
                                                  bh* __restrict__ y, int n) {
    int i = blockIdx.x * 256 + threadIdx.x;
    if (i < n) y[i] = f2bf(x[i]);
}

// h += sigmoid(t12) * t3
__global__ __launch_bounds__(256) void inject_kernel(float* __restrict__ h,
                                                     const float* __restrict__ t12,
                                                     const float* __restrict__ t3, int n) {
    int i = blockIdx.x * 256 + threadIdx.x;
    if (i < n) {
        float g = 1.0f / (1.0f + __expf(-t12[i]));
        h[i] += g * t3[i];
    }
}

// ---------------------------------------------------------------------------
// LayerNorm: fp32 rows -> bf16 rows (one wave32 per row)
// ---------------------------------------------------------------------------
__global__ __launch_bounds__(256) void ln_kernel(const float* __restrict__ x,
                                                 const float* __restrict__ w,
                                                 const float* __restrict__ b,
                                                 bh* __restrict__ y) {
    int wave = threadIdx.x >> 5, lane = threadIdx.x & 31;
    size_t row = (size_t)blockIdx.x * 8 + wave;
    const float* xr = x + row * 768;
    float s = 0.f, s2 = 0.f;
    for (int c = lane; c < 768; c += 32) {
        float v = xr[c];
        s += v; s2 += v * v;
    }
#pragma unroll
    for (int o = 16; o > 0; o >>= 1) {
        s += __shfl_xor(s, o, 32);
        s2 += __shfl_xor(s2, o, 32);
    }
    float m = s * (1.0f / 768.0f);
    float var = s2 * (1.0f / 768.0f) - m * m;
    float rs = rsqrtf(var + 1e-5f);
    bh* yr = y + row * 768;
    for (int c = lane; c < 768; c += 32)
        yr[c] = f2bf((xr[c] - m) * rs * w[c] + b[c]);
}

// ---------------------------------------------------------------------------
// bf16 GEMM: C[M,N] = A[M,K] (row-major bf16) * Wt[N,K]^T + bias, epilogues:
// MODE 0: C fp32 store; 1: C fp32 +=; 2: C bf16 store; 3: GELU -> bf16
// Block 256 thr = 8 waves (4Mx2N), tile 128x128, BK=32.
// ---------------------------------------------------------------------------
template <int MODE>
__global__ __launch_bounds__(256) void gemm_bf16(const bh* __restrict__ A,
                                                 const bh* __restrict__ B,
                                                 const float* __restrict__ bias,
                                                 void* __restrict__ Cv,
                                                 int M, int N, int K) {
    __shared__ __align__(16) bh As[128][40];
    __shared__ __align__(16) bh Bs[128][40];
    const int tid = threadIdx.x;
    const int wave = tid >> 5, lane = tid & 31;
    const int half = lane >> 4, l16 = lane & 15;
    const int waveM = wave >> 1, waveN = wave & 1;
    const size_t m0 = (size_t)blockIdx.y * 128;
    const size_t n0 = (size_t)blockIdx.x * 128;

    const int lrow = tid >> 1;
    const int lko = (tid & 1) * 16;

    const bh* Ap = A + (m0 + lrow) * K + lko;
    const bh* Bp = B + (n0 + lrow) * K + lko;

    v8f acc[2][4];
#pragma unroll
    for (int i = 0; i < 2; ++i)
#pragma unroll
        for (int j = 0; j < 4; ++j) acc[i][j] = vzero8();

    const int nk = K >> 5;
    for (int kt = 0; kt < nk; ++kt) {
        const int kb = kt * 32;
        uint4 a0 = *reinterpret_cast<const uint4*>(Ap + kb);
        uint4 a1 = *reinterpret_cast<const uint4*>(Ap + kb + 8);
        uint4 b0 = *reinterpret_cast<const uint4*>(Bp + kb);
        uint4 b1 = *reinterpret_cast<const uint4*>(Bp + kb + 8);
        if (kt + 1 < nk) {  // pull next tile toward cache while WMMAs run
            __builtin_prefetch(Ap + kb + 32, 0, 1);
            __builtin_prefetch(Bp + kb + 32, 0, 1);
        }
        __syncthreads();
        *reinterpret_cast<uint4*>(&As[lrow][lko]) = a0;
        *reinterpret_cast<uint4*>(&As[lrow][lko + 8]) = a1;
        *reinterpret_cast<uint4*>(&Bs[lrow][lko]) = b0;
        *reinterpret_cast<uint4*>(&Bs[lrow][lko + 8]) = b1;
        __syncthreads();

        Frag16 af[2], bf[4];
#pragma unroll
        for (int i = 0; i < 2; ++i) {
            const int r = waveM * 32 + i * 16 + l16;
            ld_frag(af[i], &As[r][half * 8], &As[r][16 + half * 8]);
        }
#pragma unroll
        for (int j = 0; j < 4; ++j) {
            const int r = waveN * 64 + j * 16 + l16;
            ld_frag(bf[j], &Bs[r][half * 8], &Bs[r][16 + half * 8]);
        }
#pragma unroll
        for (int i = 0; i < 2; ++i)
#pragma unroll
            for (int j = 0; j < 4; ++j)
                acc[i][j] = wmma_bf16(af[i].v, bf[j].v, acc[i][j]);
    }

    float* Cf = (float*)Cv;
    bh* Cb = (bh*)Cv;
#pragma unroll
    for (int j = 0; j < 4; ++j) {
        const int n = (int)n0 + waveN * 64 + j * 16 + l16;
        const float bn = bias[n];
#pragma unroll
        for (int i = 0; i < 2; ++i) {
#pragma unroll
            for (int r = 0; r < 8; ++r) {
                const size_t m = m0 + waveM * 32 + i * 16 + r + 8 * half;
                const float v = acc[i][j][r] + bn;
                const size_t idx = m * (size_t)N + n;
                if (MODE == 0) Cf[idx] = v;
                else if (MODE == 1) Cf[idx] += v;
                else if (MODE == 2) Cb[idx] = f2bf(v);
                else {
                    float g = 0.5f * v *
                        (1.0f + tanhf(0.7978845608028654f * (v + 0.044715f * v * v * v)));
                    Cb[idx] = f2bf(g);
                }
            }
        }
    }
}

// ---------------------------------------------------------------------------
// Flash attention: per block one (b, head, 64 q-rows). 4 waves x 16 q-rows.
// qkv bf16 layout [B*T, 3*768]; output bf16 [B*T, 768].
// Softmax reductions are pipelined: xor-offset outer loop, 8 rows in flight.
// ---------------------------------------------------------------------------
__global__ __launch_bounds__(128) void attn_kernel(const bh* __restrict__ qkv,
                                                   bh* __restrict__ xo) {
    __shared__ __align__(16) bh Qs[64][72];
    __shared__ __align__(16) bh Ks[32][72];
    __shared__ __align__(16) bh Vt[64][40];
    __shared__ __align__(16) bh Ps[4][16][40];

    const int tid = threadIdx.x;
    const int w = tid >> 5, lane = tid & 31;
    const int half = lane >> 4, l16 = lane & 15;
    const int q0 = blockIdx.x * 64;
    const int hh = blockIdx.y;
    const int b = blockIdx.z;
    const size_t rstride = 2304;
    const bh* base = qkv + (size_t)b * 1024 * rstride;

    // Q tile 64x64
    for (int c = tid; c < 512; c += 128) {
        int r = c >> 3, ch = c & 7;
        uint4 qv = *reinterpret_cast<const uint4*>(
            base + (size_t)(q0 + r) * rstride + hh * 64 + ch * 8);
        *reinterpret_cast<uint4*>(&Qs[r][ch * 8]) = qv;
    }
    __syncthreads();

    Frag16 qa[2];
#pragma unroll
    for (int d2 = 0; d2 < 2; ++d2) {
        const int r = w * 16 + l16;
        ld_frag(qa[d2], &Qs[r][d2 * 32 + half * 8], &Qs[r][d2 * 32 + 16 + half * 8]);
    }

    v8f oacc[4];
#pragma unroll
    for (int j = 0; j < 4; ++j) oacc[j] = vzero8();
    float rowm[8], rowl[8];
#pragma unroll
    for (int r = 0; r < 8; ++r) { rowm[r] = -1e30f; rowl[r] = 0.f; }

    const int nkt = q0 / 32 + 2;
    for (int kt = 0; kt < nkt; ++kt) {
        __syncthreads();
        // K tile 32x64
        for (int c = tid; c < 256; c += 128) {
            int r = c >> 3, ch = c & 7;
            uint4 kv = *reinterpret_cast<const uint4*>(
                base + (size_t)(kt * 32 + r) * rstride + 768 + hh * 64 + ch * 8);
            *reinterpret_cast<uint4*>(&Ks[r][ch * 8]) = kv;
        }
        // V tile 32x64 -> transposed Vt[64][32]
        {
            const int key = tid >> 2;
#pragma unroll
            for (int p = 0; p < 2; ++p) {
                const int cb = (tid & 3) + p * 4;
                union { uint4 u; bh e[8]; } cv;
                cv.u = *reinterpret_cast<const uint4*>(
                    base + (size_t)(kt * 32 + key) * rstride + 1536 + hh * 64 + cb * 8);
#pragma unroll
                for (int e = 0; e < 8; ++e) Vt[cb * 8 + e][key] = cv.e[e];
            }
        }
        __syncthreads();

        // S = Q K^T  (two 16x16 key tiles, Dh=64 = 2 x K32)
        v8f s0 = vzero8(), s1 = vzero8();
#pragma unroll
        for (int d2 = 0; d2 < 2; ++d2) {
            Frag16 kb0, kb1;
            ld_frag(kb0, &Ks[l16][d2 * 32 + half * 8], &Ks[l16][d2 * 32 + 16 + half * 8]);
            ld_frag(kb1, &Ks[16 + l16][d2 * 32 + half * 8],
                    &Ks[16 + l16][d2 * 32 + 16 + half * 8]);
            s0 = wmma_bf16(qa[d2].v, kb0.v, s0);
            s1 = wmma_bf16(qa[d2].v, kb1.v, s1);
        }

        // --- online softmax, pipelined across the 8 rows each lane holds ---
        const int qrow_base = q0 + w * 16 + 8 * half;
        const int k0i = kt * 32 + l16;
        float v0a[8], v1a[8], mx[8], sc[8], ps[8];
#pragma unroll
        for (int r = 0; r < 8; ++r) {
            const int qrow = qrow_base + r;
            float v0 = s0[r] * 0.125f; if (k0i > qrow) v0 = -1e30f;
            float v1 = s1[r] * 0.125f; if (k0i + 16 > qrow) v1 = -1e30f;
            v0a[r] = v0; v1a[r] = v1;
            mx[r] = fmaxf(v0, v1);
        }
        // max-reduce: 8 independent ds_bpermute chains in flight per step
#pragma unroll
        for (int o = 8; o > 0; o >>= 1) {
#pragma unroll
            for (int r = 0; r < 8; ++r)
                mx[r] = fmaxf(mx[r], __shfl_xor(mx[r], o, 32));
        }
#pragma unroll
        for (int r = 0; r < 8; ++r) {
            const float mnew = fmaxf(rowm[r], mx[r]);
            sc[r] = __expf(rowm[r] - mnew);
            rowm[r] = mnew;
            const float p0 = __expf(v0a[r] - mnew);
            const float p1 = __expf(v1a[r] - mnew);
            ps[r] = p0 + p1;
            Ps[w][r + 8 * half][l16] = f2bf(p0);
            Ps[w][r + 8 * half][16 + l16] = f2bf(p1);
#pragma unroll
            for (int jo = 0; jo < 4; ++jo) oacc[jo][r] *= sc[r];
        }
        // sum-reduce, same pipelined structure
#pragma unroll
        for (int o = 8; o > 0; o >>= 1) {
#pragma unroll
            for (int r = 0; r < 8; ++r)
                ps[r] += __shfl_xor(ps[r], o, 32);
        }
#pragma unroll
        for (int r = 0; r < 8; ++r) rowl[r] = rowl[r] * sc[r] + ps[r];

        __syncthreads();

        // O += P V
        Frag16 pf;
        ld_frag(pf, &Ps[w][l16][half * 8], &Ps[w][l16][16 + half * 8]);
#pragma unroll
        for (int jo = 0; jo < 4; ++jo) {
            Frag16 vb;
            ld_frag(vb, &Vt[jo * 16 + l16][half * 8], &Vt[jo * 16 + l16][16 + half * 8]);
            oacc[jo] = wmma_bf16(pf.v, vb.v, oacc[jo]);
        }
    }

    float rinv[8];
#pragma unroll
    for (int r = 0; r < 8; ++r) rinv[r] = 1.0f / rowl[r];
#pragma unroll
    for (int jo = 0; jo < 4; ++jo) {
#pragma unroll
        for (int r = 0; r < 8; ++r) {
            const float ov = oacc[jo][r] * rinv[r];
            const int m = q0 + w * 16 + r + 8 * half;
            xo[((size_t)b * 1024 + m) * 768 + hh * 64 + jo * 16 + l16] = f2bf(ov);
        }
    }
}

// ---------------------------------------------------------------------------
// Final LN + 60-way head on last token of each batch row (tiny).
// ---------------------------------------------------------------------------
__global__ __launch_bounds__(256) void head_kernel(const float* __restrict__ h,
                                                   const float* __restrict__ lnw,
                                                   const float* __restrict__ lnb,
                                                   const float* __restrict__ hw,
                                                   const float* __restrict__ hb2,
                                                   float* __restrict__ out) {
    __shared__ float xr[768];
    __shared__ float red[256];
    const int b = blockIdx.x, tid = threadIdx.x;
    const float* row = h + ((size_t)b * 1024 + 1023) * 768;
    float s = 0.f, s2 = 0.f;
    for (int c = tid; c < 768; c += 256) {
        float v = row[c];
        xr[c] = v; s += v; s2 += v * v;
    }
    red[tid] = s; __syncthreads();
    for (int st = 128; st; st >>= 1) { if (tid < st) red[tid] += red[tid + st]; __syncthreads(); }
    const float mean = red[0] * (1.0f / 768.0f);
    __syncthreads();
    red[tid] = s2; __syncthreads();
    for (int st = 128; st; st >>= 1) { if (tid < st) red[tid] += red[tid + st]; __syncthreads(); }
    const float var = red[0] * (1.0f / 768.0f) - mean * mean;
    const float rstd = rsqrtf(var + 1e-5f);
    __syncthreads();
    for (int c = tid; c < 768; c += 256) xr[c] = (xr[c] - mean) * rstd * lnw[c] + lnb[c];
    __syncthreads();
    if (tid < 60) {
        float acc = hb2[tid];
        for (int k = 0; k < 768; ++k) acc += xr[k] * hw[k * 60 + tid];
        out[b * 60 + tid] = acc;
    }
}

// ---------------------------------------------------------------------------
// Host orchestration
// ---------------------------------------------------------------------------
static void launch_gemm(int mode, const bh* A, const bh* B, const float* bias, void* C,
                        int M, int N, int K, hipStream_t s) {
    dim3 g(N / 128, M / 128);
    switch (mode) {
        case 0: gemm_bf16<0><<<g, 256, 0, s>>>(A, B, bias, C, M, N, K); break;
        case 1: gemm_bf16<1><<<g, 256, 0, s>>>(A, B, bias, C, M, N, K); break;
        case 2: gemm_bf16<2><<<g, 256, 0, s>>>(A, B, bias, C, M, N, K); break;
        default: gemm_bf16<3><<<g, 256, 0, s>>>(A, B, bias, C, M, N, K); break;
    }
}

static void launch_conv(const float* in, bh* out, int K, int N, hipStream_t s) {
    dim3 g(N / 32, K / 32);
    conv_w_kernel<<<g, 256, 0, s>>>(in, out, K, N);
}

extern "C" void kernel_launch(void* const* d_in, const int* in_sizes, int n_in,
                              void* d_out, int out_size, void* d_ws, size_t ws_size,
                              hipStream_t stream) {
    (void)in_sizes; (void)n_in; (void)out_size; (void)ws_size;
    const int M = 8192, D = 768, D3 = 2304, DFF = 3072;

    const int* ids = (const int*)d_in[0];
    const int* mul = (const int*)d_in[1];
    const float* wpe = (const float*)d_in[2];
    const float* lnf_w = (const float*)d_in[3];
    const float* lnf_b = (const float*)d_in[4];
    const int TB = 5 + 12 * 12;  // 149
    const float* tok_emb = (const float*)d_in[TB + 0];
    const float* state_emb = (const float*)d_in[TB + 1];
    const float* sp_w = (const float*)d_in[TB + 2];
    const float* sp_b = (const float*)d_in[TB + 3];
    const float* wh_w = (const float*)d_in[TB + 4];
    const float* wh_b = (const float*)d_in[TB + 5];
    const float* ws_w = (const float*)d_in[TB + 6];
    const float* ws_b = (const float*)d_in[TB + 7];
    const float* wd_w = (const float*)d_in[TB + 8];
    const float* wd_b = (const float*)d_in[TB + 9];
    const float* head_w = (const float*)d_in[TB + 10];
    const float* head_b = (const float*)d_in[TB + 11];

    // workspace carve (all sizes multiples of 256B)
    char* ws = (char*)d_ws;
    size_t off = 0;
    auto carve = [&](size_t bytes) { void* p = ws + off; off += (bytes + 255) & ~(size_t)255; return p; };
    bh* wt_qkv = (bh*)carve((size_t)D3 * D * 2);
    bh* wt_attnp = (bh*)carve((size_t)D * D * 2);
    bh* wt_fc = (bh*)carve((size_t)DFF * D * 2);
    bh* wt_proj = (bh*)carve((size_t)D * DFF * 2);
    bh* wt_sp = (bh*)carve((size_t)D * 128 * 2);
    bh* wt_wh = (bh*)carve((size_t)D * D * 2);
    bh* wt_ws = (bh*)carve((size_t)D * D * 2);
    bh* wt_wd = (bh*)carve((size_t)D * D * 2);
    float* h = (float*)carve((size_t)M * D * 4);
    bh* sb = (bh*)carve((size_t)M * D * 2);
    bh* xa = (bh*)carve((size_t)M * D * 2);
    bh* qkvb = (bh*)carve((size_t)M * D3 * 2);
    bh* xo = (bh*)carve((size_t)M * D * 2);
    bh* ff = (bh*)carve((size_t)M * DFF * 2);
    bh* seg = (bh*)carve((size_t)M * 128 * 2);
    int* sid = (int*)carve((size_t)M * 4);
    // layer-8 temporaries alias buffers that are dead at that point:
    float* t12 = (float*)qkvb;   // 25.2MB < 37.7MB
    float* t3 = (float*)ff;      // 25.2MB < 50.3MB
    bh* hb = xo;                 // h in bf16, xo dead by then

    const int nD = M * D;

    // 1) state scan + state embedding path
    scan_kernel<<<1, 32, 0, stream>>>(ids, mul, sid);
    launch_conv(sp_w, wt_sp, 128, D, stream);
    launch_conv(wh_w, wt_wh, D, D, stream);
    launch_conv(ws_w, wt_ws, D, D, stream);
    launch_conv(wd_w, wt_wd, D, D, stream);
    seg_kernel<<<(M * 128) / 256, 256, 0, stream>>>(sid, state_emb, seg);
    launch_gemm(2, seg, wt_sp, sp_b, sb, M, D, 128, stream);  // s -> bf16

    // 2) token + position embedding
    embed_kernel<<<M, 256, 0, stream>>>(ids, tok_emb, wpe, h);

    // 3) transformer layers
    for (int L = 0; L < 12; ++L) {
        const int base = 5 + L * 12;
        const float* ln1w = (const float*)d_in[base + 0];
        const float* ln1b = (const float*)d_in[base + 1];
        const float* attn_w = (const float*)d_in[base + 2];
        const float* attn_b = (const float*)d_in[base + 3];
        const float* attnp_w = (const float*)d_in[base + 4];
        const float* attnp_b = (const float*)d_in[base + 5];
        const float* ln2w = (const float*)d_in[base + 6];
        const float* ln2b = (const float*)d_in[base + 7];
        const float* fc_w = (const float*)d_in[base + 8];
        const float* fc_b = (const float*)d_in[base + 9];
        const float* proj_w = (const float*)d_in[base + 10];
        const float* proj_b = (const float*)d_in[base + 11];

        launch_conv(attn_w, wt_qkv, D, D3, stream);
        launch_conv(attnp_w, wt_attnp, D, D, stream);
        launch_conv(fc_w, wt_fc, D, DFF, stream);
        launch_conv(proj_w, wt_proj, DFF, D, stream);

        ln_kernel<<<M / 8, 256, 0, stream>>>(h, ln1w, ln1b, xa);
        launch_gemm(2, xa, wt_qkv, attn_b, qkvb, M, D3, D, stream);
        attn_kernel<<<dim3(16, 12, 8), 128, 0, stream>>>(qkvb, xo);
        launch_gemm(1, xo, wt_attnp, attnp_b, h, M, D, D, stream);  // h += attnproj

        ln_kernel<<<M / 8, 256, 0, stream>>>(h, ln2w, ln2b, xa);
        launch_gemm(3, xa, wt_fc, fc_b, ff, M, DFF, D, stream);     // gelu -> bf16
        launch_gemm(1, ff, wt_proj, proj_b, h, M, D, DFF, stream);  // h += mlp

        if (L == 8) {
            cvt_kernel<<<(nD + 255) / 256, 256, 0, stream>>>(h, hb, nD);
            launch_gemm(0, hb, wt_wh, wh_b, t12, M, D, D, stream);  // t12 = h Wh + bh
            launch_gemm(1, sb, wt_ws, ws_b, t12, M, D, D, stream);  // t12 += s Ws + bs
            launch_gemm(0, sb, wt_wd, wd_b, t3, M, D, D, stream);   // t3 = s Wd + bd
            inject_kernel<<<(nD + 255) / 256, 256, 0, stream>>>(h, t12, t3, nD);
        }
    }

    // 4) final LN + head (last token only)
    head_kernel<<<8, 256, 0, stream>>>(h, lnf_w, lnf_b, head_w, head_b, (float*)d_out);
}
